// GroupedQueryAttention_84954453115470
// MI455X (gfx1250) — matrix operand order repaired and checked by reading, not compiled, not run on hardware
//
#include <hip/hip_runtime.h>

// ---------------------------------------------------------------------------
// GQA for MI455X (gfx1250, wave32). GEMMs via v_wmma_f32_16x16x32_bf16,
// GEMM A-tile staging via global_load_async_to_lds_b128 (ASYNCcnt),
// K/V tile streaming via Tensor Data Mover (tensor_load_to_lds, 6-arg form),
// double-buffered with s_wait_tensorcnt.
// ---------------------------------------------------------------------------

typedef __attribute__((ext_vector_type(16))) __bf16          v16bf;
typedef __attribute__((ext_vector_type(8)))  float           v8f;
typedef __attribute__((ext_vector_type(8)))  unsigned short  usvec8;
typedef __attribute__((ext_vector_type(4)))  unsigned int    u32x4;
typedef __attribute__((ext_vector_type(8)))  int             i32x8;
typedef __attribute__((ext_vector_type(4)))  int             i32x4;

union FragBf {
    v16bf          v;
    usvec8         h[2];
    unsigned short u[16];
};

constexpr int S_  = 2048;   // sequence length
constexpr int HD_ = 64;     // head dim

__device__ __forceinline__ unsigned short f2bf(float x) {
    union { float f; unsigned u; } p; p.f = x;
    unsigned r = p.u + 0x7FFFu + ((p.u >> 16) & 1u);   // RNE
    return (unsigned short)(r >> 16);
}
__device__ __forceinline__ float bf2f(unsigned short hbits) {
    union { unsigned u; float f; } p; p.u = ((unsigned)hbits) << 16;
    return p.f;
}
__device__ __forceinline__ v8f v8zero() {
    v8f z;
#pragma unroll
    for (int i = 0; i < 8; ++i) z[i] = 0.0f;
    return z;
}
__device__ __forceinline__ v8f wmma_bf16(const FragBf& a, const FragBf& b, v8f c) {
    return __builtin_amdgcn_wmma_f32_16x16x32_bf16(
        false, a.v, false, b.v, (short)0, c, false, false);
}

#if __has_builtin(__builtin_amdgcn_tensor_load_to_lds) && \
    __has_builtin(__builtin_amdgcn_s_wait_tensorcnt)
#define HAVE_TDM 1
#else
#define HAVE_TDM 0
#endif

#if __has_builtin(__builtin_amdgcn_global_load_async_to_lds_b128) && \
    __has_builtin(__builtin_amdgcn_s_wait_asynccnt)
#define HAVE_ASYNC 1
#else
#define HAVE_ASYNC 0
#endif

#if HAVE_TDM
// 2D TDM load: tile_dim0=64 elems (bf16), tile_dim1=rows, tensor row stride 64.
// This toolchain exposes the 6-arg builtin: (g0, g1, g2, g3, g4, cpol).
__device__ __forceinline__ void tdm_load_2d(unsigned lds_byte_off,
                                            unsigned long long gaddr,
                                            int tensor_rows, int tile_rows) {
    u32x4 g0; i32x8 g1; i32x4 g2; i32x4 g3; i32x8 g4;
    g0[0] = 1u;                                              // count=1, user mode
    g0[1] = lds_byte_off;                                    // lds_addr
    g0[2] = (unsigned)(gaddr & 0xFFFFFFFFull);               // global_addr[31:0]
    g0[3] = (unsigned)((gaddr >> 32) & 0x1FFFFFFull) | (2u << 30); // [56:32] | type=2
    g1[0] = (int)(1u << 16);                                 // data_size = 2 bytes
    g1[1] = (int)(64u << 16);                                // tensor_dim0 = 64 (bits 79:48)
    g1[2] = (int)(((unsigned)tensor_rows & 0xFFFFu) << 16);  // tensor_dim1 lo (bits 111:80)
    g1[3] = (int)((((unsigned)tensor_rows >> 16) & 0xFFFFu) | (64u << 16)); // tile_dim0=64
    g1[4] = (int)((unsigned)tile_rows & 0xFFFFu);            // tile_dim1, tile_dim2=0
    g1[5] = 64;                                              // tensor_dim0_stride (bits 207:160)
    g1[6] = 0; g1[7] = 0;                                    // tensor_dim1_stride unused (2D)
#pragma unroll
    for (int i = 0; i < 4; ++i) { g2[i] = 0; g3[i] = 0; }
#pragma unroll
    for (int i = 0; i < 8; ++i) g4[i] = 0;
    __builtin_amdgcn_tensor_load_to_lds(g0, g1, g2, g3, g4, 0);
}
__device__ __forceinline__ unsigned lds_off(const void* p) {
    return (unsigned)(unsigned long long)(size_t)p;          // generic addr low 32 = LDS offset
}
#endif

// ---------------------------------------------------------------------------
// f32 -> bf16 raw conversion (grid-stride)
// ---------------------------------------------------------------------------
__global__ void cvt_f32_bf16(const float* __restrict__ in,
                             unsigned short* __restrict__ out, size_t n) {
    size_t i = (size_t)blockIdx.x * blockDim.x + threadIdx.x;
    size_t stride = (size_t)gridDim.x * blockDim.x;
    for (; i < n; i += stride) out[i] = f2bf(in[i]);
}

// ---------------------------------------------------------------------------
// Tiled bf16 WMMA GEMM: C[M,N] = A[M,K] * B[K,N], row-major bf16 operands.
// Block tile 128x64 (4 waves, each 32x64 -> 8 WMMAs per K-step of 32).
//   mode 0: scatter bf16 into attention layout [(b*Hout+h)*S + s]*hd + d
//   mode 1: write f32 row-major
// ---------------------------------------------------------------------------
__global__ __launch_bounds__(128) void gemm_bf16_wmma(
    const unsigned short* __restrict__ A,
    const unsigned short* __restrict__ B,
    void* __restrict__ Cout,
    int N, int K, int mode, int Hout)
{
    __shared__ __align__(16) unsigned short As[128 * 32];   // [m][k]
    __shared__ __align__(16) unsigned short Bts[64 * 32];   // transposed [n][k]

    const int tid  = threadIdx.x;
    const int wave = tid >> 5;
    const int lane = tid & 31;
    const int m0 = blockIdx.y * 128;
    const int n0 = blockIdx.x * 64;

    v8f acc[2][4];
#pragma unroll
    for (int f = 0; f < 2; ++f)
#pragma unroll
        for (int t = 0; t < 4; ++t) acc[f][t] = v8zero();

    // --- strength-reduced staging pointers
    const int ar = tid >> 2;                 // A stage row 0..31 (strided by 32)
    const int ac = (tid & 3) * 8;            // A stage col chunk
    const unsigned short* pa = A + (size_t)(m0 + ar) * K + ac;
    const size_t aRowStride = (size_t)32 * K;

    const int bk = tid >> 3;                 // B stage k-row 0..15 (strided by 16)
    const int bc = (tid & 7) * 8;            // B stage n chunk
    const unsigned short* pb = B + (size_t)bk * N + n0 + bc;
    const size_t bRowStride = (size_t)16 * N;
    const size_t bStep = (size_t)32 * N;

    const int mrow0 = wave * 32 + (lane & 15);
    const int kbA = (lane >> 4) * 8;         // A-frag chunk base
    const int kbB = (lane >> 4) * 16;        // B-frag K base

    for (int k0 = 0; k0 < K; k0 += 32) {
        // ---- A tile: direct global -> LDS async copies (no VGPR bounce)
#if HAVE_ASYNC
#pragma unroll
        for (int q = 0; q < 4; ++q)
            __builtin_amdgcn_global_load_async_to_lds_b128(
                (__attribute__((address_space(1))) i32x4*)(pa + (size_t)q * aRowStride),
                (__attribute__((address_space(3))) i32x4*)&As[(ar + 32 * q) * 32 + ac],
                0, 0);
#else
        {
            usvec8 aTmp[4];
#pragma unroll
            for (int q = 0; q < 4; ++q)
                aTmp[q] = *(const usvec8*)(pa + (size_t)q * aRowStride);
#pragma unroll
            for (int q = 0; q < 4; ++q)
                *(usvec8*)&As[(ar + 32 * q) * 32 + ac] = aTmp[q];
        }
#endif
        // ---- B tile: batched loads, then transpose scatter into Bts[n][k]
        {
            usvec8 bTmp[2];
#pragma unroll
            for (int q = 0; q < 2; ++q)
                bTmp[q] = *(const usvec8*)(pb + (size_t)q * bRowStride);
#pragma unroll
            for (int q = 0; q < 2; ++q)
#pragma unroll
                for (int j = 0; j < 8; ++j)
                    Bts[(bc + j) * 32 + bk + 16 * q] = bTmp[q][j];
        }
        pa += 32;
        pb += bStep;
        if (k0 + 32 < K) __builtin_prefetch(pa, 0, 1);
#if HAVE_ASYNC
        __builtin_amdgcn_s_wait_asynccnt(0);
#endif
        __syncthreads();

        FragBf bfg[4];
#pragma unroll
        for (int t = 0; t < 4; ++t) {
            int ncol = t * 16 + (lane & 15);
            bfg[t].h[0] = *(const usvec8*)&Bts[ncol * 32 + kbB];
            bfg[t].h[1] = *(const usvec8*)&Bts[ncol * 32 + kbB + 8];
        }
#pragma unroll
        for (int f = 0; f < 2; ++f) {
            FragBf af;
            int mr = mrow0 + f * 16;
            af.h[0] = *(const usvec8*)&As[mr * 32 + kbB];
            af.h[1] = *(const usvec8*)&As[mr * 32 + kbB + 8];
#pragma unroll
            for (int t = 0; t < 4; ++t)
                acc[f][t] = wmma_bf16(af, bfg[t], acc[f][t]);
        }
        __syncthreads();
    }

    // --- epilogue (C layout: rows in VGPR index, cols in lanes)
#pragma unroll
    for (int f = 0; f < 2; ++f)
#pragma unroll
    for (int t = 0; t < 4; ++t)
#pragma unroll
    for (int j = 0; j < 8; ++j) {
        int r = (lane < 16) ? j : j + 8;
        int m = m0 + wave * 32 + f * 16 + r;
        int n = n0 + t * 16 + (lane & 15);
        float val = acc[f][t][j];
        if (mode == 0) {
            int b = m >> 11, s = m & (S_ - 1);
            int hh = n >> 6, d = n & (HD_ - 1);
            ((unsigned short*)Cout)[(((size_t)b * Hout + hh) * S_ + s) * HD_ + d] = f2bf(val);
        } else {
            ((float*)Cout)[(size_t)m * N + n] = val;
        }
    }
}

// ---------------------------------------------------------------------------
// RoPE in-place on bf16 [rows][64]; row % S gives sequence position.
// ---------------------------------------------------------------------------
__global__ void rope_bf16(unsigned short* __restrict__ buf,
                          const float* __restrict__ cosb,
                          const float* __restrict__ sinb,
                          size_t nrows)
{
    size_t idx = (size_t)blockIdx.x * blockDim.x + threadIdx.x;
    if (idx >= nrows * 32) return;
    int d = (int)(idx & 31);
    size_t row = idx >> 5;
    int s = (int)(row & (size_t)(S_ - 1));
    unsigned short* p = buf + row * 64;
    float x1 = bf2f(p[d]);
    float x2 = bf2f(p[d + 32]);
    float c1 = cosb[s * 64 + d],      s1 = sinb[s * 64 + d];
    float c2 = cosb[s * 64 + d + 32], s2 = sinb[s * 64 + d + 32];
    p[d]      = f2bf(x1 * c1 - x2 * s1);
    p[d + 32] = f2bf(x2 * c2 + x1 * s2);
}

// ---------------------------------------------------------------------------
// Flash attention: one wave per (b, h, 16-query tile). 32-key blocks.
// K/V tiles streamed by TDM (double-buffered, s_wait_tensorcnt) when present.
// ---------------------------------------------------------------------------
__global__ __launch_bounds__(32) void flash_attn_wmma(
    const unsigned short* __restrict__ Q,
    const unsigned short* __restrict__ Kbuf,
    const unsigned short* __restrict__ Vbuf,
    unsigned short* __restrict__ Ctx,
    int B, int H, int G, float scale)
{
    __shared__ __align__(16) unsigned short Ks[2][32 * 64];
    __shared__ __align__(16) unsigned short Vs[2][32 * 64];
    __shared__ __align__(16) unsigned short Ps[16 * 32];

    const int bh = blockIdx.y;
    const int b = bh / H, h = bh - b * H;
    const int g = h / (H / G);
    const int q0 = blockIdx.x * 16;
    const int lane = threadIdx.x;
    const int nlo = lane & 15;
    const int kbA = (lane >> 4) * 8;
    const int kbB = (lane >> 4) * 16;

    const unsigned short* Qp = Q    + ((size_t)(b * H + h) * S_) * 64;
    const unsigned short* Kp = Kbuf + ((size_t)(b * G + g) * S_) * 64;
    const unsigned short* Vp = Vbuf + ((size_t)(b * G + g) * S_) * 64;

    FragBf qf[2];
#pragma unroll
    for (int c = 0; c < 2; ++c) {
        qf[c].h[0] = *(const usvec8*)&Qp[(size_t)(q0 + nlo) * 64 + c * 32 + kbB];
        qf[c].h[1] = *(const usvec8*)&Qp[(size_t)(q0 + nlo) * 64 + c * 32 + kbB + 8];
    }

    float mrun[8], lrun[8];
#pragma unroll
    for (int j = 0; j < 8; ++j) { mrun[j] = -3.0e38f; lrun[j] = 0.0f; }
    v8f ctx[4];
#pragma unroll
    for (int t = 0; t < 4; ++t) ctx[t] = v8zero();

#if HAVE_TDM
    const unsigned long long kG = (unsigned long long)(size_t)Kp;
    const unsigned long long vG = (unsigned long long)(size_t)Vp;
    const unsigned ksO[2] = { lds_off(&Ks[0][0]), lds_off(&Ks[1][0]) };
    const unsigned vsO[2] = { lds_off(&Vs[0][0]), lds_off(&Vs[1][0]) };
    // prologue: tile 0 into buffer 0
    tdm_load_2d(ksO[0], kG, S_, 32);
    tdm_load_2d(vsO[0], vG, S_, 32);
#endif

    for (int kt = 0; kt < S_; kt += 32) {
#if HAVE_TDM
        const int cur = (kt >> 5) & 1;
        if (kt + 32 < S_) {
            unsigned long long off = (unsigned long long)(kt + 32) * 64ull * 2ull;
            tdm_load_2d(ksO[cur ^ 1], kG + off, S_, 32);
            tdm_load_2d(vsO[cur ^ 1], vG + off, S_, 32);
            __builtin_amdgcn_s_wait_tensorcnt((short)2);   // previous pair done
        } else {
            __builtin_amdgcn_s_wait_tensorcnt((short)0);
        }
        __syncthreads();                                    // compiler fence (1-wave WG)
#else
        const int cur = 0;
#pragma unroll
        for (int q = 0; q < 8; ++q) {
            int i = lane + q * 32;
            int r = i >> 3, c = (i & 7) * 8;
            *(usvec8*)&Ks[0][r * 64 + c] = *(const usvec8*)&Kp[(size_t)(kt + r) * 64 + c];
            *(usvec8*)&Vs[0][r * 64 + c] = *(const usvec8*)&Vp[(size_t)(kt + r) * 64 + c];
        }
        __syncthreads();
#endif
        const unsigned short* Kst = &Ks[cur][0];
        const unsigned short* Vst = &Vs[cur][0];

        // ---- scores (4 WMMAs): s0 = keys 0-15, s1 = keys 16-31
        v8f s0 = v8zero(), s1 = v8zero();
#pragma unroll
        for (int c = 0; c < 2; ++c) {
            int db = kbB + c * 32;
            FragBf b0, b1;
            b0.h[0] = *(const usvec8*)&Kst[nlo * 64 + db];
            b0.h[1] = *(const usvec8*)&Kst[nlo * 64 + db + 8];
            b1.h[0] = *(const usvec8*)&Kst[(nlo + 16) * 64 + db];
            b1.h[1] = *(const usvec8*)&Kst[(nlo + 16) * 64 + db + 8];
            s0 = wmma_bf16(qf[c], b0, s0);
            s1 = wmma_bf16(qf[c], b1, s1);
        }

        // ---- online softmax (row stats reduced across 16-lane halves)
        float rmax[8];
#pragma unroll
        for (int j = 0; j < 8; ++j) {
            s0[j] *= scale; s1[j] *= scale;
            rmax[j] = fmaxf(s0[j], s1[j]);
        }
#pragma unroll
        for (int msk = 1; msk < 16; msk <<= 1)
#pragma unroll
            for (int j = 0; j < 8; ++j)
                rmax[j] = fmaxf(rmax[j], __shfl_xor(rmax[j], msk, 32));

        float alpha[8], p0[8], p1[8], rsum[8];
#pragma unroll
        for (int j = 0; j < 8; ++j) {
            float mnew = fmaxf(mrun[j], rmax[j]);
            alpha[j] = __expf(mrun[j] - mnew);
            mrun[j] = mnew;
            p0[j] = __expf(s0[j] - mnew);
            p1[j] = __expf(s1[j] - mnew);
            rsum[j] = p0[j] + p1[j];
        }
#pragma unroll
        for (int msk = 1; msk < 16; msk <<= 1)
#pragma unroll
            for (int j = 0; j < 8; ++j)
                rsum[j] += __shfl_xor(rsum[j], msk, 32);
#pragma unroll
        for (int j = 0; j < 8; ++j) lrun[j] = lrun[j] * alpha[j] + rsum[j];
#pragma unroll
        for (int t = 0; t < 4; ++t)
#pragma unroll
            for (int j = 0; j < 8; ++j) ctx[t][j] *= alpha[j];

        // ---- P (C layout) -> LDS -> A-fragment layout
        {
            int rbase = (lane >> 4) * 8;
#pragma unroll
            for (int j = 0; j < 8; ++j) {
                Ps[(rbase + j) * 32 + nlo]      = f2bf(p0[j]);
                Ps[(rbase + j) * 32 + nlo + 16] = f2bf(p1[j]);
            }
        }
        __syncthreads();
        FragBf pf;
        pf.h[0] = *(const usvec8*)&Ps[nlo * 32 + kbB];
        pf.h[1] = *(const usvec8*)&Ps[nlo * 32 + kbB + 8];

        // ---- ctx += P * V  (4 WMMAs over 4 d-tiles)
#pragma unroll
        for (int t = 0; t < 4; ++t) {
            FragBf vf;
            int d = t * 16 + nlo;
#pragma unroll
            for (int j = 0; j < 16; ++j) vf.u[j] = Vst[(kbB + j) * 64 + d];
            ctx[t] = wmma_bf16(pf, vf, ctx[t]);
        }
        __syncthreads();
    }

    // ---- finalize: ctx layout [B][S][H*64]
#pragma unroll
    for (int t = 0; t < 4; ++t) {
#pragma unroll
        for (int j = 0; j < 8; ++j) {
            int r = (lane < 16) ? j : j + 8;
            int q = q0 + r;
            float val = ctx[t][j] / lrun[j];
            Ctx[((size_t)(b * S_ + q) * H + h) * 64 + t * 16 + nlo] = f2bf(val);
        }
    }
}

// ---------------------------------------------------------------------------
// Launcher
// ---------------------------------------------------------------------------
extern "C" void kernel_launch(void* const* d_in, const int* in_sizes, int n_in,
                              void* d_out, int out_size, void* d_ws, size_t ws_size,
                              hipStream_t stream)
{
    constexpr int B = 2, D = 2048, H = 32, G = 8;
    constexpr int M = B * S_;                // 4096
    constexpr int Nq = H * HD_;              // 2048
    constexpr int Nkv = G * HD_;             // 512

    const float* x    = (const float*)d_in[0];
    const float* cosb = (const float*)d_in[2];
    const float* sinb = (const float*)d_in[3];
    const float* Wq   = (const float*)d_in[4];
    const float* Wk   = (const float*)d_in[5];
    const float* Wv   = (const float*)d_in[6];
    const float* Wo   = (const float*)d_in[7];

    unsigned short* ws = (unsigned short*)d_ws;
    unsigned short* Xb   = ws;
    unsigned short* Wqb  = Xb  + (size_t)M * D;
    unsigned short* Wkb  = Wqb + (size_t)D * Nq;
    unsigned short* Wvb  = Wkb + (size_t)D * Nkv;
    unsigned short* Wob  = Wvb + (size_t)D * Nkv;
    unsigned short* Qb   = Wob + (size_t)Nq * Nq;
    unsigned short* Kb   = Qb  + (size_t)B * H * S_ * HD_;
    unsigned short* Vb   = Kb  + (size_t)B * G * S_ * HD_;
    unsigned short* Ctxb = Vb  + (size_t)B * G * S_ * HD_;

    cvt_f32_bf16<<<2048, 256, 0, stream>>>(x,  Xb,  (size_t)M * D);
    cvt_f32_bf16<<<2048, 256, 0, stream>>>(Wq, Wqb, (size_t)D * Nq);
    cvt_f32_bf16<<<1024, 256, 0, stream>>>(Wk, Wkb, (size_t)D * Nkv);
    cvt_f32_bf16<<<1024, 256, 0, stream>>>(Wv, Wvb, (size_t)D * Nkv);
    cvt_f32_bf16<<<2048, 256, 0, stream>>>(Wo, Wob, (size_t)Nq * Nq);

    gemm_bf16_wmma<<<dim3(Nq / 64,  M / 128), 128, 0, stream>>>(
        Xb, Wqb, Qb, Nq,  D, /*mode=*/0, H);
    gemm_bf16_wmma<<<dim3(Nkv / 64, M / 128), 128, 0, stream>>>(
        Xb, Wkb, Kb, Nkv, D, /*mode=*/0, G);
    gemm_bf16_wmma<<<dim3(Nkv / 64, M / 128), 128, 0, stream>>>(
        Xb, Wvb, Vb, Nkv, D, /*mode=*/0, G);

    {
        size_t qrows = (size_t)B * H * S_;
        size_t krows = (size_t)B * G * S_;
        rope_bf16<<<(unsigned)((qrows * 32 + 255) / 256), 256, 0, stream>>>(Qb, cosb, sinb, qrows);
        rope_bf16<<<(unsigned)((krows * 32 + 255) / 256), 256, 0, stream>>>(Kb, cosb, sinb, krows);
    }

    flash_attn_wmma<<<dim3(S_ / 16, B * H), 32, 0, stream>>>(
        Qb, Kb, Vb, Ctxb, B, H, G, 0.125f);

    gemm_bf16_wmma<<<dim3(Nq / 64, M / 128), 128, 0, stream>>>(
        Ctxb, Wob, d_out, Nq, Nq, /*mode=*/1, H);

    (void)in_sizes; (void)n_in; (void)out_size; (void)ws_size;
}
